// TalkingHeadAttention_11871289606671
// MI455X (gfx1250) — matrix operand order repaired and checked by reading, not compiled, not run on hardware
//
#include <hip/hip_runtime.h>

// Talking-head attention, MI455X (gfx1250, wave32, WMMA bf16 16x16x32).
// B=4, N=1024, DIM=768, H=12, HD=64, SCALE=0.125.

#define BB   4
#define NN   1024
#define DIMK 768
#define HH   12
#define HDK  64
#define SCALEF 0.125f

typedef __attribute__((ext_vector_type(16))) __bf16 v16bf_t;
typedef __attribute__((ext_vector_type(8)))  float  v8f_t;

__device__ __forceinline__ __bf16 f2bf(float f) {
  union { float f; unsigned u; } x; x.f = f;
  unsigned r = (x.u + 0x7FFFu + ((x.u >> 16) & 1u)) >> 16;
  union { unsigned short s; __bf16 b; } y; y.s = (unsigned short)r;
  return y.b;
}

__device__ __forceinline__ v8f_t wmma_bf16(v16bf_t a, v16bf_t b, v8f_t c) {
  // emits v_wmma_f32_16x16x32_bf16
  return __builtin_amdgcn_wmma_f32_16x16x32_bf16(false, a, false, b, (short)0, c, false, false);
}

// A fragment (16x32 bf16, row-major source, stride ld elements).
// lane L<16: row m0+L, K = {kh..kh+7, kh+16..kh+23} with kh=0; lanes 16-31: kh=8.
__device__ __forceinline__ v16bf_t load_a_bf16(const __bf16* A, int ld, int m0, int k0) {
  int lane = threadIdx.x & 31;
  const __bf16* p = A + (size_t)(m0 + (lane & 15)) * ld + k0 + ((lane >> 4) << 3);
  union { v16bf_t v; float4 q[2]; } u;
  u.q[0] = *reinterpret_cast<const float4*>(p);
  u.q[1] = *reinterpret_cast<const float4*>(p + 16);
  return u.v;
}

// A fragment from an f32 source (convert to bf16 on load).
__device__ __forceinline__ v16bf_t load_a_f32(const float* A, int ld, int m0, int k0) {
  int lane = threadIdx.x & 31;
  const float* p = A + (size_t)(m0 + (lane & 15)) * ld + k0 + ((lane >> 4) << 3);
  float4 f0 = *reinterpret_cast<const float4*>(p);
  float4 f1 = *reinterpret_cast<const float4*>(p + 4);
  float4 f2 = *reinterpret_cast<const float4*>(p + 16);
  float4 f3 = *reinterpret_cast<const float4*>(p + 20);
  union { v16bf_t v; __bf16 e[16]; } u;
  u.e[0] = f2bf(f0.x); u.e[1] = f2bf(f0.y); u.e[2]  = f2bf(f0.z); u.e[3]  = f2bf(f0.w);
  u.e[4] = f2bf(f1.x); u.e[5] = f2bf(f1.y); u.e[6]  = f2bf(f1.z); u.e[7]  = f2bf(f1.w);
  u.e[8] = f2bf(f2.x); u.e[9] = f2bf(f2.y); u.e[10] = f2bf(f2.z); u.e[11] = f2bf(f2.w);
  u.e[12]= f2bf(f3.x); u.e[13]= f2bf(f3.y); u.e[14] = f2bf(f3.z); u.e[15] = f2bf(f3.w);
  return u.v;
}

// B fragment (32x16 bf16) from a TRANSPOSED operand BT (row = output column,
// stride ld along K): lane L holds column n0+(L&15), K = {kh..kh+7, kh+16..kh+23} (kh = (L>>4)*8).
__device__ __forceinline__ v16bf_t load_b_bf16T(const __bf16* BT, int ld, int n0, int k0) {
  int lane = threadIdx.x & 31;
  const __bf16* p = BT + (size_t)(n0 + (lane & 15)) * ld + k0 + ((lane >> 4) << 3);
  union { v16bf_t v; float4 q[2]; } u;
  u.q[0] = *reinterpret_cast<const float4*>(p);
  u.q[1] = *reinterpret_cast<const float4*>(p + 16);
  return u.v;
}

// ---------------- K0a: f32 -> bf16 elementwise ----------------
__global__ void convert_bf16_kernel(const float* __restrict__ src, __bf16* __restrict__ dst, int n) {
  int i = blockIdx.x * blockDim.x + threadIdx.x;
  if (i < n) dst[i] = f2bf(src[i]);
}

// ---------------- K0b: f32 [K,N] -> bf16 transposed [N,K] ----------------
__global__ void transpose_bf16_kernel(const float* __restrict__ src, __bf16* __restrict__ dst,
                                      int K, int N) {
  int i = blockIdx.x * blockDim.x + threadIdx.x;
  if (i < K * N) {
    int n = i / K, k = i - n * K;
    dst[i] = f2bf(src[(size_t)k * N + n]);
  }
}

// ---------------- K1: QKV GEMM [4096,768]x[768,2304], 32x64/wave, scatter Q/K/Vt ----------------
__global__ __launch_bounds__(256) void qkv_gemm_kernel(
    const __bf16* __restrict__ xb, const __bf16* __restrict__ wT,
    const float* __restrict__ bias,
    __bf16* __restrict__ Qb, __bf16* __restrict__ Kb, __bf16* __restrict__ Vtb) {
  int wave = (blockIdx.x * blockDim.x + threadIdx.x) >> 5;
  int lane = threadIdx.x & 31;
  const int NG = (3 * DIMK) / 64;           // 36 column groups of 64
  int mt = wave / NG, ng = wave - mt * NG;
  if (mt >= (BB * NN) / 32) return;         // wave-uniform
  int m0 = mt * 32, n0 = ng * 64;

  v8f_t acc[2][4];
  for (int u = 0; u < 2; ++u)
    for (int t = 0; t < 4; ++t)
      for (int e = 0; e < 8; ++e) acc[u][t][e] = 0.f;

  for (int k0 = 0; k0 < DIMK; k0 += 32) {
    if (k0 + 64 < DIMK)
      __builtin_prefetch(xb + (size_t)(m0 + (lane & 15)) * DIMK + k0 + 64, 0, 1);
    v16bf_t a0 = load_a_bf16(xb, DIMK, m0, k0);
    v16bf_t a1 = load_a_bf16(xb, DIMK, m0 + 16, k0);
#pragma unroll
    for (int t = 0; t < 4; ++t) {
      v16bf_t bfr = load_b_bf16T(wT, DIMK, n0 + t * 16, k0);
      acc[0][t] = wmma_bf16(a0, bfr, acc[0][t]);
      acc[1][t] = wmma_bf16(a1, bfr, acc[1][t]);
    }
  }

  int cl = lane & 15, rh = (lane >> 4) << 3;
#pragma unroll
  for (int t = 0; t < 4; ++t) {
    int c = n0 + t * 16 + cl;
    int which = c / DIMK;
    int within = c - which * DIMK;
    int h = within >> 6, d = within & 63;
    float bv = bias[c];
#pragma unroll
    for (int u = 0; u < 2; ++u) {
#pragma unroll
      for (int v = 0; v < 8; ++v) {
        int m = m0 + u * 16 + v + rh;
        int b = m >> 10, n = m & 1023;
        float val = acc[u][t][v] + bv;
        size_t bh = (size_t)(b * HH + h);
        if (which == 0)      Qb[(bh * NN + n) * HDK + d] = f2bf(val * SCALEF);
        else if (which == 1) Kb[(bh * NN + n) * HDK + d] = f2bf(val);
        else                 Vtb[(bh * HDK + d) * NN + n] = f2bf(val);
      }
    }
  }
}

// ---------------- K2: S = Q K^T per (b,h), 32x64/wave ----------------
__global__ __launch_bounds__(256) void scores_kernel(
    const __bf16* __restrict__ Qb, const __bf16* __restrict__ Kb, float* __restrict__ S) {
  int wave = (blockIdx.x * blockDim.x + threadIdx.x) >> 5;
  int lane = threadIdx.x & 31;
  int jg = wave & 15, it = (wave >> 4) & 31, bh = wave >> 9;
  if (bh >= BB * HH) return;
  const __bf16* Q = Qb + (size_t)bh * NN * HDK;
  const __bf16* K = Kb + (size_t)bh * NN * HDK;
  float* Sp = S + ((size_t)bh << 20);
  int i0 = it * 32, j0 = jg * 64;

  v8f_t acc[2][4];
  for (int u = 0; u < 2; ++u)
    for (int t = 0; t < 4; ++t)
      for (int e = 0; e < 8; ++e) acc[u][t][e] = 0.f;

  for (int k0 = 0; k0 < HDK; k0 += 32) {
    v16bf_t a0 = load_a_bf16(Q, HDK, i0, k0);
    v16bf_t a1 = load_a_bf16(Q, HDK, i0 + 16, k0);
#pragma unroll
    for (int t = 0; t < 4; ++t) {
      v16bf_t bfr = load_b_bf16T(K, HDK, j0 + t * 16, k0);
      acc[0][t] = wmma_bf16(a0, bfr, acc[0][t]);
      acc[1][t] = wmma_bf16(a1, bfr, acc[1][t]);
    }
  }
  int cl = lane & 15, rh = (lane >> 4) << 3;
#pragma unroll
  for (int t = 0; t < 4; ++t) {
    int j = j0 + t * 16 + cl;
#pragma unroll
    for (int u = 0; u < 2; ++u)
#pragma unroll
      for (int v = 0; v < 8; ++v)
        Sp[(size_t)(i0 + u * 16 + v + rh) * NN + j] = acc[u][t][v];
  }
}

// ---------------- K3: head-mix -> softmax -> head-mix, in place on S ----------------
__global__ __launch_bounds__(256) void mix_softmax_kernel(
    float* __restrict__ S,
    const float* __restrict__ wl, const float* __restrict__ bl,
    const float* __restrict__ ww, const float* __restrict__ bw) {
  __shared__ float sm[HH][NN];
  __shared__ float red[256];
  __shared__ float wlS[HH * HH], wwS[HH * HH], blS[HH], bwS[HH];

  int tid = threadIdx.x;
  int b = blockIdx.x >> 10, i = blockIdx.x & 1023;

  if (tid < HH * HH) { wlS[tid] = wl[tid]; wwS[tid] = ww[tid]; }
  if (tid < HH)      { blS[tid] = bl[tid]; bwS[tid] = bw[tid]; }

  for (int h = 0; h < HH; ++h) {
    size_t base = (((size_t)(b * HH + h)) * NN + i) * NN;
    for (int j = tid; j < NN; j += 256) sm[h][j] = S[base + j];
  }
  __syncthreads();

  // pre-softmax mix: t[g] = sum_h s[h] * wl[h][g] + bl[g]   (in place, register staged)
  float vals[4][HH];
  for (int jj = 0; jj < 4; ++jj)
    for (int h = 0; h < HH; ++h) vals[jj][h] = sm[h][tid + jj * 256];
  __syncthreads();
  for (int jj = 0; jj < 4; ++jj)
    for (int g = 0; g < HH; ++g) {
      float t = blS[g];
      for (int h = 0; h < HH; ++h) t += vals[jj][h] * wlS[h * HH + g];
      sm[g][tid + jj * 256] = t;
    }
  __syncthreads();

  // softmax over j for each head row g
  for (int g = 0; g < HH; ++g) {
    float pm = -1e30f;
    for (int jj = 0; jj < 4; ++jj) pm = fmaxf(pm, sm[g][tid + jj * 256]);
    red[tid] = pm; __syncthreads();
    for (int s = 128; s > 0; s >>= 1) {
      if (tid < s) red[tid] = fmaxf(red[tid], red[tid + s]);
      __syncthreads();
    }
    float m = red[0]; __syncthreads();
    float ps = 0.f;
    for (int jj = 0; jj < 4; ++jj) {
      float e = __expf(sm[g][tid + jj * 256] - m);
      sm[g][tid + jj * 256] = e;
      ps += e;
    }
    red[tid] = ps; __syncthreads();
    for (int s = 128; s > 0; s >>= 1) {
      if (tid < s) red[tid] += red[tid + s];
      __syncthreads();
    }
    float inv = 1.f / red[0]; __syncthreads();
    for (int jj = 0; jj < 4; ++jj) sm[g][tid + jj * 256] *= inv;
    __syncthreads();
  }

  // post-softmax mix: p2[g] = sum_h p[h] * ww[h][g] + bw[g]
  for (int jj = 0; jj < 4; ++jj)
    for (int h = 0; h < HH; ++h) vals[jj][h] = sm[h][tid + jj * 256];
  __syncthreads();
  for (int jj = 0; jj < 4; ++jj)
    for (int g = 0; g < HH; ++g) {
      float t = bwS[g];
      for (int h = 0; h < HH; ++h) t += vals[jj][h] * wwS[h * HH + g];
      sm[g][tid + jj * 256] = t;
    }
  __syncthreads();

  for (int h = 0; h < HH; ++h) {
    size_t base = (((size_t)(b * HH + h)) * NN + i) * NN;
    for (int j = tid; j < NN; j += 256) S[base + j] = sm[h][j];
  }
}

// ---------------- K4: O = P V per (b,h), 16x64/wave, write [B,N,H,HD] bf16 ----------------
__global__ __launch_bounds__(256) void pv_kernel(
    const float* __restrict__ S, const __bf16* __restrict__ Vtb, __bf16* __restrict__ Obuf) {
  int wave = (blockIdx.x * blockDim.x + threadIdx.x) >> 5;
  int lane = threadIdx.x & 31;
  int it = wave & 63, bh = wave >> 6;
  if (bh >= BB * HH) return;
  const float* P = S + ((size_t)bh << 20);
  const __bf16* VT = Vtb + (size_t)bh * HDK * NN;
  int i0 = it * 16;

  v8f_t acc[4];
  for (int t = 0; t < 4; ++t) for (int e = 0; e < 8; ++e) acc[t][e] = 0.f;

  for (int k0 = 0; k0 < NN; k0 += 32) {
    if (k0 + 64 < NN)
      __builtin_prefetch(P + (size_t)(i0 + (lane & 15)) * NN + k0 + 64, 0, 1);
    v16bf_t a = load_a_f32(P, NN, i0, k0);
#pragma unroll
    for (int t = 0; t < 4; ++t)
      acc[t] = wmma_bf16(a, load_b_bf16T(VT, NN, t * 16, k0), acc[t]);
  }
  int b = bh / HH, h = bh - b * HH;
  int cl = lane & 15, rh = (lane >> 4) << 3;
#pragma unroll
  for (int t = 0; t < 4; ++t) {
    int d = t * 16 + cl;
#pragma unroll
    for (int v = 0; v < 8; ++v) {
      int i = i0 + v + rh;
      Obuf[((size_t)(b * NN + i)) * DIMK + h * HDK + d] = f2bf(acc[t][v]);
    }
  }
}

// ---------------- K5: out = O @ w_proj + b_proj (f32 output), 32x64/wave ----------------
__global__ __launch_bounds__(256) void proj_kernel(
    const __bf16* __restrict__ Obuf, const __bf16* __restrict__ wT,
    const float* __restrict__ bias, float* __restrict__ out) {
  int wave = (blockIdx.x * blockDim.x + threadIdx.x) >> 5;
  int lane = threadIdx.x & 31;
  const int NG = DIMK / 64;                 // 12
  int mt = wave / NG, ng = wave - mt * NG;
  if (mt >= (BB * NN) / 32) return;
  int m0 = mt * 32, n0 = ng * 64;

  v8f_t acc[2][4];
  for (int u = 0; u < 2; ++u)
    for (int t = 0; t < 4; ++t)
      for (int e = 0; e < 8; ++e) acc[u][t][e] = 0.f;

  for (int k0 = 0; k0 < DIMK; k0 += 32) {
    if (k0 + 64 < DIMK)
      __builtin_prefetch(Obuf + (size_t)(m0 + (lane & 15)) * DIMK + k0 + 64, 0, 1);
    v16bf_t a0 = load_a_bf16(Obuf, DIMK, m0, k0);
    v16bf_t a1 = load_a_bf16(Obuf, DIMK, m0 + 16, k0);
#pragma unroll
    for (int t = 0; t < 4; ++t) {
      v16bf_t bfr = load_b_bf16T(wT, DIMK, n0 + t * 16, k0);
      acc[0][t] = wmma_bf16(a0, bfr, acc[0][t]);
      acc[1][t] = wmma_bf16(a1, bfr, acc[1][t]);
    }
  }
  int cl = lane & 15, rh = (lane >> 4) << 3;
#pragma unroll
  for (int t = 0; t < 4; ++t) {
    int c = n0 + t * 16 + cl;
    float bv = bias[c];
#pragma unroll
    for (int u = 0; u < 2; ++u)
#pragma unroll
      for (int v = 0; v < 8; ++v)
        out[(size_t)(m0 + u * 16 + v + rh) * DIMK + c] = acc[u][t][v] + bv;
  }
}

extern "C" void kernel_launch(void* const* d_in, const int* in_sizes, int n_in,
                              void* d_out, int out_size, void* d_ws, size_t ws_size,
                              hipStream_t stream) {
  const float* x      = (const float*)d_in[0];
  const float* w_qkv  = (const float*)d_in[1];
  const float* b_qkv  = (const float*)d_in[2];
  const float* w_l    = (const float*)d_in[3];
  const float* b_l    = (const float*)d_in[4];
  const float* w_w    = (const float*)d_in[5];
  const float* b_w    = (const float*)d_in[6];
  const float* w_proj = (const float*)d_in[7];
  const float* b_proj = (const float*)d_in[8];
  float* out = (float*)d_out;

  char* ws = (char*)d_ws;
  size_t off = 0;
  auto take = [&](size_t bytes) -> char* {
    char* p = ws + off;
    off += (bytes + 255) & ~(size_t)255;
    return p;
  };
  const size_t MROWS = (size_t)BB * NN;                 // 4096
  __bf16* xb     = (__bf16*)take(MROWS * DIMK * 2);     // x in bf16
  __bf16* wqkvT  = (__bf16*)take((size_t)3 * DIMK * DIMK * 2); // [2304,768]
  __bf16* wprojT = (__bf16*)take((size_t)DIMK * DIMK * 2);     // [768,768]
  __bf16* Qb     = (__bf16*)take((size_t)BB * HH * NN * HDK * 2);
  __bf16* Kb     = (__bf16*)take((size_t)BB * HH * NN * HDK * 2);
  __bf16* Vtb    = (__bf16*)take((size_t)BB * HH * HDK * NN * 2);
  float*  S      = (float*) take((size_t)BB * HH * NN * NN * 4); // 201 MB
  __bf16* Obuf   = (__bf16*)take(MROWS * DIMK * 2);

  // K0: conversions
  {
    int n = (int)(MROWS * DIMK);
    convert_bf16_kernel<<<(n + 255) / 256, 256, 0, stream>>>(x, xb, n);
  }
  transpose_bf16_kernel<<<(DIMK * 3 * DIMK + 255) / 256, 256, 0, stream>>>(w_qkv, wqkvT, DIMK, 3 * DIMK);
  transpose_bf16_kernel<<<(DIMK * DIMK + 255) / 256, 256, 0, stream>>>(w_proj, wprojT, DIMK, DIMK);

  // K1: QKV GEMM -> Q (scaled) / K / V^T, head-major.  128 m-tiles x 36 n-groups
  qkv_gemm_kernel<<<(128 * 36) / 8, 256, 0, stream>>>(xb, wqkvT, b_qkv, Qb, Kb, Vtb);

  // K2: S = Q K^T.  48 bh x 32 i-tiles x 16 j-groups
  scores_kernel<<<(48 * 32 * 16) / 8, 256, 0, stream>>>(Qb, Kb, S);

  // K3: talking-head mix -> softmax -> mix (in place on S)
  mix_softmax_kernel<<<BB * NN, 256, 0, stream>>>(S, w_l, b_l, w_w, b_w);

  // K4: O = P V.  48 bh x 64 i-tiles
  pv_kernel<<<(48 * 64) / 8, 256, 0, stream>>>(S, Vtb, Obuf);

  // K5: out = O @ w_proj + b_proj.  128 m-tiles x 12 n-groups
  proj_kernel<<<(128 * 12) / 8, 256, 0, stream>>>(Obuf, wprojT, b_proj, out);
}
